// GNNLayer_27101243638465
// MI455X (gfx1250) — compile-verified, matrix operand order checked by reading, not run on hardware
//
#include <hip/hip_runtime.h>
#include <hip/hip_bf16.h>

typedef __attribute__((ext_vector_type(16))) _Float16 v16h;
typedef __attribute__((ext_vector_type(8)))  _Float16 v8h;
typedef __attribute__((ext_vector_type(8)))  float    v8f;

#define LEAKY_SLOPE 0.01f

__device__ __forceinline__ float elu_f(float x)   { return x > 0.f ? x : (expf(x) - 1.f); }
__device__ __forceinline__ float leaky_f(float x) { return x > 0.f ? x : LEAKY_SLOPE * x; }

// ---------------------------------------------------------------------------
// WMMA GEMM:  C[M x Nn] = actOut( actIn(A[M x K]) @ B[K x Nn] + bias )
// A fp32 row-major (lda=K).  BhT is B pre-converted to f16 and TRANSPOSED to
// [Nn][K] so each lane's B fragment is two contiguous b128 loads per K-step.
// One wave owns a 16 x (16*NT) strip: the A fragment (4x b128 + 16 cvt) is
// amortized over NT wmma issues per 32-K step.
// ---------------------------------------------------------------------------
template <int ACT_IN, int ACT_OUT, int NT>
__global__ __launch_bounds__(256)
void wmma_gemm(const float* __restrict__ A, const _Float16* __restrict__ BhT,
               const float* __restrict__ bias, float* __restrict__ C,
               int Mrows, int K, int Nn)
{
    const int mtiles  = (Mrows + 15) >> 4;
    const int nstrips = Nn / (16 * NT);
    const int wave    = blockIdx.x * (blockDim.x >> 5) + (threadIdx.x >> 5);
    if (wave >= mtiles * nstrips) return;        // wave-uniform: EXEC all-ones at wmma
    const int mt   = wave / nstrips;
    const int ns   = wave % nstrips;
    const int lane = threadIdx.x & 31;
    const int hl   = lane >> 4;                  // half-wave select
    const int lrow = lane & 15;

    // A lane mapping (16-bit A 16x32): row = lane&15, kbase = (lane>>4)*8,
    // elems 0..7 -> K[kb..kb+7], elems 8..15 -> K[kb+16..kb+23]
    int arow = mt * 16 + lrow;
    if (arow >= Mrows) arow = Mrows - 1;         // tail tiles read a safe row
    const float* Abase = A + (size_t)arow * K;

    // B lane mapping (16-bit B 32x16): col = lane&15, same K split as A.
    const _Float16* Bcol[NT];
    #pragma unroll
    for (int t = 0; t < NT; ++t)
        Bcol[t] = BhT + (size_t)((ns * NT + t) * 16 + lrow) * K;

    v8f acc[NT];
    #pragma unroll
    for (int t = 0; t < NT; ++t) acc[t] = (v8f){};

    for (int kk = 0; kk < K; kk += 32) {
        const int kb = kk + hl * 8;
        __builtin_prefetch(Abase + kk + 32, 0, 0);         // global_prefetch_b8

        const float4* ap = (const float4*)(Abase + kb);    // 32B aligned (kb%8==0)
        float4 a0 = ap[0], a1 = ap[1];                     // K[kb..kb+7]
        float4 a2 = ap[4], a3 = ap[5];                     // K[kb+16..kb+23]
        float af[16] = { a0.x, a0.y, a0.z, a0.w, a1.x, a1.y, a1.z, a1.w,
                         a2.x, a2.y, a2.z, a2.w, a3.x, a3.y, a3.z, a3.w };
        v16h a;
        #pragma unroll
        for (int i = 0; i < 16; ++i) {
            float v = af[i];
            if (ACT_IN) v = elu_f(v);
            a[i] = (_Float16)v;
        }

        #pragma unroll
        for (int t = 0; t < NT; ++t) {
            v8h b0 = *(const v8h*)(Bcol[t] + kb);          // b128
            v8h b1 = *(const v8h*)(Bcol[t] + kb + 16);     // b128
            v16h b = __builtin_shufflevector(b0, b1, 0, 1, 2, 3, 4, 5, 6, 7,
                                             8, 9, 10, 11, 12, 13, 14, 15);
            // (neg_a, A, neg_b, B, c_mod, C, reuse_a, reuse_b)
            acc[t] = __builtin_amdgcn_wmma_f32_16x16x32_f16(false, a, false, b,
                                                            (short)0, acc[t],
                                                            false, false);
        }
    }

    const bool fullM = (mt * 16 + 16) <= Mrows;            // wave-uniform
    #pragma unroll
    for (int t = 0; t < NT; ++t) {
        const int col = (ns * NT + t) * 16 + lrow;
        const float bv = bias ? bias[col] : 0.f;
        float* Cb = C + (size_t)(mt * 16 + hl * 8) * Nn + col;
        if (fullM) {
            #pragma unroll
            for (int r = 0; r < 8; ++r) {                  // C: VGPR r -> M=r+8*hl
                float v = acc[t][r] + bv;
                if (ACT_OUT) v = leaky_f(v);
                Cb[(size_t)r * Nn] = v;
            }
        } else {
            #pragma unroll
            for (int r = 0; r < 8; ++r) {
                const int orow = mt * 16 + hl * 8 + r;
                if (orow < Mrows) {
                    float v = acc[t][r] + bv;
                    if (ACT_OUT) v = leaky_f(v);
                    Cb[(size_t)r * Nn] = v;
                }
            }
        }
    }
}

// ---------------------------------------------------------------------------
// Graph / utility kernels
// ---------------------------------------------------------------------------
__global__ void k_fill1(float* p, int n)
{
    int i = blockIdx.x * blockDim.x + threadIdx.x;
    if (i < n) p[i] = 1.0f;                                // self-loop contributes 1
}

__global__ void k_deg(const int* __restrict__ dst, float* deg, int E)
{
    int e = blockIdx.x * blockDim.x + threadIdx.x;
    if (e < E) atomicAdd(&deg[dst[e]], 1.0f);
}

__global__ void k_rsqrt_inplace(float* p, int n)
{
    int i = blockIdx.x * blockDim.x + threadIdx.x;
    if (i < n) p[i] = rsqrtf(p[i]);
}

// f32 [K x Nn] -> f16 transposed [Nn x K]
__global__ void k_f16T(const float* __restrict__ s, _Float16* __restrict__ d,
                       int K, int Nn)
{
    int i = blockIdx.x * blockDim.x + threadIdx.x;
    if (i < K * Nn) {
        int k = i / Nn, n = i % Nn;
        d[(size_t)n * K + k] = (_Float16)s[i];
    }
}

// agg[i][c] = h[i][c] * dis[i]^2   (self-loop term), CH=128, float4 per thread
__global__ void k_selfinit(const float* __restrict__ h, const float* __restrict__ dis,
                           float* __restrict__ agg, int n)
{
    int i = blockIdx.x * blockDim.x + threadIdx.x;
    int node = i >> 5;                                     // 128/4 = 32 chunks/node
    int c    = (i & 31) * 4;
    if (node < n) {
        float s = dis[node]; s *= s;
        float4 v = *(const float4*)(h + (size_t)node * 128 + c);
        v.x *= s; v.y *= s; v.z *= s; v.w *= s;
        *(float4*)(agg + (size_t)node * 128 + c) = v;
    }
}

// agg[dst] += h[src] * dis[src]*dis[dst]   (float atomics, L2-resident)
__global__ void k_scatter(const int* __restrict__ srcI, const int* __restrict__ dstI,
                          const float* __restrict__ dis, const float* __restrict__ h,
                          float* __restrict__ agg, int E)
{
    int gid = blockIdx.x * blockDim.x + threadIdx.x;
    int e = gid >> 5;
    int c = (gid & 31) * 4;
    if (e < E) {
        int s = srcI[e], d = dstI[e];
        float w = dis[s] * dis[d];
        float4 v = *(const float4*)(h + (size_t)s * 128 + c);
        float* o = agg + (size_t)d * 128 + c;
        atomicAdd(o + 0, v.x * w);
        atomicAdd(o + 1, v.y * w);
        atomicAdd(o + 2, v.z * w);
        atomicAdd(o + 3, v.w * w);
    }
}

// out[m] = dot(tf[ts[m][0]], tg[ts[m][1]])  over 32 channels
__global__ void k_pairdot(const int* __restrict__ ts, const float* __restrict__ tf,
                          const float* __restrict__ tg, float* __restrict__ out, int M)
{
    int m = blockIdx.x * blockDim.x + threadIdx.x;
    if (m < M) {
        int a = ts[2 * m + 0];
        int b = ts[2 * m + 1];
        const float4* pa = (const float4*)(tf + (size_t)a * 32);
        const float4* pb = (const float4*)(tg + (size_t)b * 32);
        float s = 0.f;
        #pragma unroll
        for (int i = 0; i < 8; ++i) {
            float4 x = pa[i], y = pb[i];
            s += x.x * y.x + x.y * y.y + x.z * y.z + x.w * y.w;
        }
        out[m] = s;
    }
}

// ---------------------------------------------------------------------------
extern "C" void kernel_launch(void* const* d_in, const int* in_sizes, int n_in,
                              void* d_out, int out_size, void* d_ws, size_t ws_size,
                              hipStream_t stream)
{
    const float* x    = (const float*)d_in[0];
    const int*   ei   = (const int*)  d_in[1];
    const int*   ts   = (const int*)  d_in[2];
    const float* W1   = (const float*)d_in[3];
    const float* W2   = (const float*)d_in[4];
    const float* tfw1 = (const float*)d_in[5];
    const float* tfb1 = (const float*)d_in[6];
    const float* tfw2 = (const float*)d_in[7];
    const float* tfb2 = (const float*)d_in[8];
    const float* tgw1 = (const float*)d_in[9];
    const float* tgb1 = (const float*)d_in[10];
    const float* tgw2 = (const float*)d_in[11];
    const float* tgb2 = (const float*)d_in[12];

    const int D_IN = 256, H1 = 128, H2 = 128, H3 = 64, D_OUT = 32;
    const int N = in_sizes[0] / D_IN;
    const int E = in_sizes[1] / 2;
    const int M = in_sizes[2] / 2;
    const int* srcI = ei;
    const int* dstI = ei + E;
    float* out = (float*)d_out;

    // workspace carve-up (256B aligned)
    char* w = (char*)d_ws;
    auto alloc = [&](size_t bytes) -> char* {
        char* p = w; w += (bytes + 255) & ~(size_t)255; return p;
    };
    float*    dis = (float*)   alloc((size_t)N * 4);
    _Float16* W1h = (_Float16*)alloc((size_t)D_IN * H1 * 2);   // all transposed [N][K]
    _Float16* W2h = (_Float16*)alloc((size_t)H1 * H2 * 2);
    _Float16* F1h = (_Float16*)alloc((size_t)H2 * H3 * 2);
    _Float16* F2h = (_Float16*)alloc((size_t)H3 * D_OUT * 2);
    _Float16* G1h = (_Float16*)alloc((size_t)H2 * H3 * 2);
    _Float16* G2h = (_Float16*)alloc((size_t)H3 * D_OUT * 2);
    float*    B1  = (float*)   alloc((size_t)N * H1 * 4);      // h1 -> h2 -> head temps
    float*    B2  = (float*)   alloc((size_t)N * H1 * 4);      // agg1 -> embed
    float*    B3  = (float*)   alloc((size_t)N * D_OUT * 4);   // tg
    (void)ws_size; (void)n_in; (void)out_size;

    const int T = 256;
    auto cdiv = [](long long a, long long b) { return (int)((a + b - 1) / b); };

    // 1) symmetric-norm degrees: deg = 1 (self loop) + #incoming, dis = rsqrt(deg)
    k_fill1<<<cdiv(N, T), T, 0, stream>>>(dis, N);
    k_deg  <<<cdiv(E, T), T, 0, stream>>>(dstI, dis, E);
    k_rsqrt_inplace<<<cdiv(N, T), T, 0, stream>>>(dis, N);

    // 2) weights -> f16, transposed to [N][K] for contiguous B fragments
    k_f16T<<<cdiv(D_IN * H1,   T), T, 0, stream>>>(W1,   W1h, D_IN, H1);
    k_f16T<<<cdiv(H1 * H2,     T), T, 0, stream>>>(W2,   W2h, H1,   H2);
    k_f16T<<<cdiv(H2 * H3,     T), T, 0, stream>>>(tfw1, F1h, H2,   H3);
    k_f16T<<<cdiv(H3 * D_OUT,  T), T, 0, stream>>>(tfw2, F2h, H3,   D_OUT);
    k_f16T<<<cdiv(H2 * H3,     T), T, 0, stream>>>(tgw1, G1h, H2,   H3);
    k_f16T<<<cdiv(H3 * D_OUT,  T), T, 0, stream>>>(tgw2, G2h, H3,   D_OUT);

    auto gemm_blocks = [&](int m, int nn, int nt) {
        return cdiv((long long)((m + 15) / 16) * (nn / (16 * nt)), 8);
    };

    // 3) GCN layer 1: h1 = x @ W1 ; agg1 = scatter(h1)
    wmma_gemm<0, 0, 4><<<gemm_blocks(N, H1, 4), T, 0, stream>>>(x, W1h, nullptr, B1, N, D_IN, H1);
    k_selfinit<<<cdiv((long long)N * 32, T), T, 0, stream>>>(B1, dis, B2, N);
    k_scatter <<<cdiv((long long)E * 32, T), T, 0, stream>>>(srcI, dstI, dis, B1, B2, E);

    // 4) GCN layer 2: h2 = elu(agg1) @ W2 ; embed = scatter(h2)
    wmma_gemm<1, 0, 4><<<gemm_blocks(N, H2, 4), T, 0, stream>>>(B2, W2h, nullptr, B1, N, H1, H2);
    k_selfinit<<<cdiv((long long)N * 32, T), T, 0, stream>>>(B1, dis, B2, N);
    k_scatter <<<cdiv((long long)E * 32, T), T, 0, stream>>>(srcI, dstI, dis, B1, B2, E);

    // 5) heads (bias + leaky fused into GEMM epilogue)
    float* t1  = B1;                      // N x 64 temp
    float* tfO = B1 + (size_t)N * H3;     // N x 32 (kept live)
    wmma_gemm<0, 1, 4><<<gemm_blocks(N, H3, 4),    T, 0, stream>>>(B2, F1h, tfb1, t1,  N, H2, H3);
    wmma_gemm<0, 1, 2><<<gemm_blocks(N, D_OUT, 2), T, 0, stream>>>(t1, F2h, tfb2, tfO, N, H3, D_OUT);
    wmma_gemm<0, 1, 4><<<gemm_blocks(N, H3, 4),    T, 0, stream>>>(B2, G1h, tgb1, t1,  N, H2, H3);
    wmma_gemm<0, 1, 2><<<gemm_blocks(N, D_OUT, 2), T, 0, stream>>>(t1, G2h, tgb2, B3,  N, H3, D_OUT);

    // 6) pairwise dot over train samples
    k_pairdot<<<cdiv(M, T), T, 0, stream>>>(ts, tfO, B3, out, M);
}
